// HGT_36155034698321
// MI455X (gfx1250) — matrix operand order repaired and checked by reading, not compiled, not run on hardware
//
#include <hip/hip_runtime.h>
#include <hip/hip_bf16.h>
#include <math.h>

typedef __attribute__((ext_vector_type(16))) _Float16 v16h;
typedef __attribute__((ext_vector_type(8)))  _Float16 v8h;
typedef __attribute__((ext_vector_type(4)))  _Float16 h4;
typedef __attribute__((ext_vector_type(8)))  float    v8f;

#define HIDN  128
#define HEADS 4
#define DHEAD 32
#define SA_STRIDE (HIDN + 8)   // 136 halves = 272 B, multiple of 16 B

__device__ __forceinline__ float gelu_tanh(float x) {
    float x3 = x * x * x;
    return 0.5f * x * (1.0f + tanhf(0.7978845608028654f * (x + 0.044715f * x3)));
}

// order-preserving float -> uint map (for atomic max over floats)
__device__ __forceinline__ unsigned int fmap(float f) {
    unsigned int b = __float_as_uint(f);
    return (b & 0x80000000u) ? ~b : (b | 0x80000000u);
}
__device__ __forceinline__ float funmap(unsigned int u) {
    unsigned int b = (u & 0x80000000u) ? (u ^ 0x80000000u) : ~u;
    return __uint_as_float(b);
}

// ---------------------------------------------------------------------------
// Typed linear: OUT[n,:] = act(X)[n,:] @ W[type(n)] + Bias[type(n)]
// Optional GELU on input; optional per-type sigmoid skip gate:
//   OUT = a*OUT + (1-a)*Hprev,  a = sigmoid(skipv[type(n)])
// ntype == nullptr -> all rows type 0 (relation GEMMs, nTypes=1).
// Block: 16 nodes x 128 outputs, 8 wave32's; wave w owns cols [16w,16w+16).
// A tile staged in LDS (f16, reused by all 8 waves); B fragments built
// directly from global (L2-resident weights, 64B-coalesced dword loads).
// K=128 as 4 chained v_wmma_f32_16x16x32_f16.
// ---------------------------------------------------------------------------
__global__ __launch_bounds__(256)
void typed_linear_wmma(const float* __restrict__ X,
                       const float* __restrict__ W,     // [T][128][128]
                       const float* __restrict__ Bias,  // [T][128] or null
                       const int*   __restrict__ ntype, // [N] or null
                       const float* __restrict__ skipv, // [T] or null
                       const float* __restrict__ Hprev, // [N][128] or null
                       float* __restrict__ OUT,         // [N][128]
                       int nNodes, int nTypes, int geluIn)
{
    __shared__ _Float16 sA[16][SA_STRIDE];

    const int tid  = threadIdx.x;
    const int lane = tid & 31;
    const int w    = tid >> 5;
    const int row0 = blockIdx.x * 16;

    // ---- stage A tile: f32 -> f16 (optional gelu), packed b64 stores ----
    for (int i = tid; i < 16 * 32; i += 256) {       // 512 float4 loads
        int r  = i >> 5;
        int c4 = (i & 31) << 2;
        int node = row0 + r;
        float4 xv = make_float4(0.f, 0.f, 0.f, 0.f);
        if (node < nNodes)
            xv = *(const float4*)(X + (size_t)node * HIDN + c4);
        if (geluIn) {
            xv.x = gelu_tanh(xv.x); xv.y = gelu_tanh(xv.y);
            xv.z = gelu_tanh(xv.z); xv.w = gelu_tanh(xv.w);
        }
        h4 hv = { (_Float16)xv.x, (_Float16)xv.y, (_Float16)xv.z, (_Float16)xv.w };
        *(h4*)&sA[r][c4] = hv;
    }
    __syncthreads();

    // ---- preload A fragments (ISA 16-bit A 16x32 layout) ----
    // lane<16: row=lane, K in {0..7} u {16..23}; lane>=16: same row, K +8.
    // Each half-chunk is 8 contiguous halves -> ds_load_b128.
    v16h afrag[4];
    {
        const int ar = lane & 15;
        const int kh = (lane >> 4) & 1;
        for (int kb = 0; kb < 4; ++kb) {
            v8h lo = *(const v8h*)&sA[ar][kb * 32 + kh * 8];
            v8h hi = *(const v8h*)&sA[ar][kb * 32 + 16 + kh * 8];
            afrag[kb] = __builtin_shufflevector(lo, hi,
                0, 1, 2, 3, 4, 5, 6, 7, 8, 9, 10, 11, 12, 13, 14, 15);
        }
    }

    const int bc = lane & 15;            // column within the 16-wide tile
    const int ks = (lane >> 4) & 1;      // K-half select
    const int colg  = w * 16 + bc;
    const int rbase = (lane >> 4) * 8;

    // hoist node types / bounds for this lane's 8 output rows
    int nt[8];
    for (int g = 0; g < 8; ++g) {
        const int node = row0 + rbase + g;
        nt[g] = (node < nNodes) ? (ntype ? ntype[node] : 0) : -1;
    }

    for (int t = 0; t < nTypes; ++t) {
        const float* Wt = W + (size_t)t * HIDN * HIDN;
        if (t + 1 < nTypes)   // global_prefetch_b8 the next type's slab
            __builtin_prefetch(Wt + HIDN * HIDN + (size_t)lane * 64, 0, 0);

        v8f acc = {};
        #pragma unroll
        for (int kb = 0; kb < 4; ++kb) {
            // B fragment straight from global: lane (bc,ks) needs
            // col = w*16+bc, K = kb*32 + ks*8 + (j<8 ? j : j+8), j=0..15.
            const float* colp = Wt + (size_t)(kb * 32 + ks * 8) * HIDN + colg;
            v16h bf;
            #pragma unroll
            for (int j = 0; j < 16; ++j)
                bf[j] = (_Float16)colp[(size_t)(j < 8 ? j : j + 8) * HIDN];
            acc = __builtin_amdgcn_wmma_f32_16x16x32_f16(
                false, afrag[kb], false, bf, (short)0, acc, false, false);
        }

        // ---- epilogue: C layout -> rows (lane>>4)*8+g, col lane&15 ----
        #pragma unroll
        for (int g = 0; g < 8; ++g) {
            if (nt[g] == t) {
                const int node = row0 + rbase + g;
                float val = acc[g];
                if (Bias) val += Bias[t * HIDN + colg];
                if (skipv) {
                    float a = 1.0f / (1.0f + __expf(-skipv[t]));
                    val = a * val + (1.0f - a) * Hprev[(size_t)node * HIDN + colg];
                }
                OUT[(size_t)node * HIDN + colg] = val;
            }
        }
    }
}

// ---------------------------------------------------------------------------
// Expand [R][H][DH][DH] relation weights into [R][128][128] block-diagonal
// matrices so the relation transform becomes a plain 128x128 WMMA GEMM.
// ---------------------------------------------------------------------------
__global__ void expand_blockdiag(const float* __restrict__ rel,
                                 float* __restrict__ bd, int Rr)
{
    int idx = blockIdx.x * blockDim.x + threadIdx.x;
    int total = Rr * HIDN * HIDN;
    if (idx >= total) return;
    int r  = idx / (HIDN * HIDN);
    int ij = idx % (HIDN * HIDN);
    int i  = ij / HIDN, j = ij % HIDN;
    int hi = i >> 5, hj = j >> 5;
    float v = 0.f;
    if (hi == hj)
        v = rel[(((size_t)r * HEADS + hi) * DHEAD + (i & 31)) * DHEAD + (j & 31)];
    bd[idx] = v;
}

// ---- edge phase ----------------------------------------------------------

__global__ void attn_init(float* __restrict__ agg, float* __restrict__ den,
                          unsigned int* __restrict__ mx, int N_)
{
    size_t idx = (size_t)blockIdx.x * blockDim.x + threadIdx.x;
    if (idx < (size_t)N_ * HIDN)  agg[idx] = 0.f;
    if (idx < (size_t)N_ * HEADS) { den[idx] = 0.f; mx[idx] = 0x007FFFFFu; } // fmap(-inf)
}

__global__ void att_logits(const float* __restrict__ q, const float* __restrict__ kt,
                           const int* __restrict__ ei, const int* __restrict__ et,
                           const float* __restrict__ mul, float* __restrict__ att,
                           int E_, int N_)
{
    size_t idx = (size_t)blockIdx.x * blockDim.x + threadIdx.x;
    if (idx >= (size_t)E_ * HEADS) return;
    int e = (int)(idx >> 2), h = (int)(idx & 3);
    int s = ei[e], d = ei[(size_t)E_ + e], r = et[e];
    const float4* qp = (const float4*)(q + (size_t)d * HIDN + h * DHEAD);
    const float4* kp = (const float4*)(kt + ((size_t)r * N_ + s) * HIDN + h * DHEAD);
    float sum = 0.f;
    for (int i = 0; i < DHEAD / 4; ++i) {
        float4 a = qp[i], b = kp[i];
        sum += a.x * b.x + a.y * b.y + a.z * b.z + a.w * b.w;
    }
    att[idx] = sum * mul[r * HEADS + h] * 0.17677669529663687f; // 1/sqrt(32)
}

__global__ void seg_max(const float* __restrict__ att, const int* __restrict__ ei,
                        unsigned int* __restrict__ mx, int E_)
{
    size_t idx = (size_t)blockIdx.x * blockDim.x + threadIdx.x;
    if (idx >= (size_t)E_ * HEADS) return;
    int e = (int)(idx >> 2), h = (int)(idx & 3);
    int d = ei[(size_t)E_ + e];
    atomicMax(&mx[(size_t)d * HEADS + h], fmap(att[idx]));
}

__global__ void seg_exp_den(const float* __restrict__ att, const int* __restrict__ ei,
                            const unsigned int* __restrict__ mx,
                            float* __restrict__ den, float* __restrict__ ex, int E_)
{
    size_t idx = (size_t)blockIdx.x * blockDim.x + threadIdx.x;
    if (idx >= (size_t)E_ * HEADS) return;
    int e = (int)(idx >> 2), h = (int)(idx & 3);
    int d = ei[(size_t)E_ + e];
    float m = funmap(mx[(size_t)d * HEADS + h]);
    float v = __expf(att[idx] - m);
    ex[idx] = v;
    atomicAdd(&den[(size_t)d * HEADS + h], v);
}

__global__ void agg_scatter(const float* __restrict__ ex, const float* __restrict__ den,
                            const float* __restrict__ vt, const int* __restrict__ ei,
                            const int* __restrict__ et, float* __restrict__ agg,
                            int E_, int N_)
{
    size_t idx = (size_t)blockIdx.x * blockDim.x + threadIdx.x;
    if (idx >= (size_t)E_ * HIDN) return;
    int e = (int)(idx >> 7);
    int c = (int)(idx & 127);
    int h = c >> 5;
    int s = ei[e], d = ei[(size_t)E_ + e], r = et[e];
    float alpha = ex[(size_t)e * HEADS + h] / (den[(size_t)d * HEADS + h] + 1e-16f);
    atomicAdd(&agg[(size_t)d * HIDN + c],
              alpha * vt[((size_t)r * N_ + s) * HIDN + c]);
}

// ---------------------------------------------------------------------------
extern "C" void kernel_launch(void* const* d_in, const int* in_sizes, int n_in,
                              void* d_out, int out_size, void* d_ws, size_t ws_size,
                              hipStream_t stream)
{
    const float* x     = (const float*)d_in[0];
    const float* W_in  = (const float*)d_in[1];
    const float* b_in  = (const float*)d_in[2];
    const float* Wk    = (const float*)d_in[3];
    const float* bk    = (const float*)d_in[4];
    const float* Wq    = (const float*)d_in[5];
    const float* bq    = (const float*)d_in[6];
    const float* Wv    = (const float*)d_in[7];
    const float* bv    = (const float*)d_in[8];
    const float* Wa    = (const float*)d_in[9];
    const float* ba    = (const float*)d_in[10];
    const float* a_rel = (const float*)d_in[11];
    const float* m_rel = (const float*)d_in[12];
    const float* mu    = (const float*)d_in[13];
    const float* skip  = (const float*)d_in[14];
    const int* ntype   = (const int*)d_in[15];
    const int* eidx    = (const int*)d_in[16];
    const int* etype   = (const int*)d_in[17];

    const int N = in_sizes[0] / HIDN;
    const int T = in_sizes[2] / HIDN;               // b_in is (T,128)
    const int E = in_sizes[17];                      // edge_type is (E,)
    const int L = in_sizes[14] / T;                  // skip is (L,T)
    const int R = in_sizes[13] / (L * HEADS);        // mu is (L,R,H)
    (void)n_in; (void)ws_size; (void)out_size;

    // ---- carve workspace ----
    const size_t NH = (size_t)N * HIDN;
    float* p    = (float*)d_ws;
    float* hbuf = p; p += NH;
    float* h2   = p; p += NH;
    float* kb   = p; p += NH;
    float* qb   = p; p += NH;
    float* vb   = p; p += NH;
    float* agg  = p; p += NH;
    float* ktb  = p; p += (size_t)R * NH;
    float* vtb  = p; p += (size_t)R * NH;
    float* attb = p; p += (size_t)E * HEADS;
    float* exb  = p; p += (size_t)E * HEADS;
    float* denb = p; p += (size_t)N * HEADS;
    unsigned int* mxb = (unsigned int*)p; p += (size_t)N * HEADS;
    float* bdA  = p; p += (size_t)R * HIDN * HIDN;
    float* bdM  = p; p += (size_t)R * HIDN * HIDN;

    const unsigned tiles   = (unsigned)((N + 15) / 16);
    const unsigned gEH     = (unsigned)(((size_t)E * HEADS + 255) / 256);
    const unsigned gNH     = (unsigned)((NH + 255) / 256);
    const unsigned gEhid   = (unsigned)(((size_t)E * HIDN + 255) / 256);
    const unsigned gBD     = (unsigned)(((size_t)R * HIDN * HIDN + 255) / 256);

    // input projection: h = typed_linear(x, W_in, b_in)
    typed_linear_wmma<<<tiles, 256, 0, stream>>>(
        x, W_in, b_in, ntype, nullptr, nullptr, hbuf, N, T, 0);

    const float* cur = hbuf;
    float* nxt = h2;

    for (int l = 0; l < L; ++l) {
        const size_t wOff = (size_t)l * T * HIDN * HIDN;
        const size_t bOff = (size_t)l * T * HIDN;

        typed_linear_wmma<<<tiles, 256, 0, stream>>>(
            cur, Wk + wOff, bk + bOff, ntype, nullptr, nullptr, kb, N, T, 0);
        typed_linear_wmma<<<tiles, 256, 0, stream>>>(
            cur, Wq + wOff, bq + bOff, ntype, nullptr, nullptr, qb, N, T, 0);
        typed_linear_wmma<<<tiles, 256, 0, stream>>>(
            cur, Wv + wOff, bv + bOff, ntype, nullptr, nullptr, vb, N, T, 0);

        // relation transforms as block-diagonal 128x128 GEMMs
        const size_t relOff = (size_t)l * R * HEADS * DHEAD * DHEAD;
        expand_blockdiag<<<gBD, 256, 0, stream>>>(a_rel + relOff, bdA, R);
        expand_blockdiag<<<gBD, 256, 0, stream>>>(m_rel + relOff, bdM, R);
        for (int r = 0; r < R; ++r) {
            typed_linear_wmma<<<tiles, 256, 0, stream>>>(
                kb, bdA + (size_t)r * HIDN * HIDN, nullptr, nullptr, nullptr,
                nullptr, ktb + (size_t)r * NH, N, 1, 0);
            typed_linear_wmma<<<tiles, 256, 0, stream>>>(
                vb, bdM + (size_t)r * HIDN * HIDN, nullptr, nullptr, nullptr,
                nullptr, vtb + (size_t)r * NH, N, 1, 0);
        }

        // edge phase: logits -> segment softmax -> weighted scatter
        attn_init<<<gNH, 256, 0, stream>>>(agg, denb, mxb, N);
        att_logits<<<gEH, 256, 0, stream>>>(
            qb, ktb, eidx, etype, mu + (size_t)l * R * HEADS, attb, E, N);
        seg_max<<<gEH, 256, 0, stream>>>(attb, eidx, mxb, E);
        seg_exp_den<<<gEH, 256, 0, stream>>>(attb, eidx, mxb, denb, exb, E);
        agg_scatter<<<gEhid, 256, 0, stream>>>(exb, denb, vtb, eidx, etype, agg, E, N);

        // output projection with gelu(agg) input + per-type skip gate
        float* outp = (l == L - 1) ? (float*)d_out : nxt;
        typed_linear_wmma<<<tiles, 256, 0, stream>>>(
            agg, Wa + wOff, ba + bOff, ntype, skip + (size_t)l * T, cur,
            outp, N, T, 1);

        if (l != L - 1) {
            const float* t0 = cur; cur = outp; nxt = (float*)t0;
        }
    }
}
